// QuantLinear_8796093022603
// MI455X (gfx1250) — matrix operand order repaired and checked by reading, not compiled, not run on hardware
//
#include <hip/hip_runtime.h>

// ---------------------------------------------------------------------------
// W4A16 group-quantized GEMM for gfx1250 (MI455X), wave32 + WMMA.
//   out[m, n] = fp16( sum_k x[m,k] * (nib(qweight,k,n) - zero(g,n)) * scale(g,n) ) + bias[n]
//   M = B*S = 8192, K = 4096, N = 11008, group = 128 (g = k >> 7)
// Double-buffered LDS pipeline, packed-fp16 magic-bias dequant.
// ---------------------------------------------------------------------------

typedef _Float16 v16h __attribute__((ext_vector_type(16)));
typedef _Float16 v8h  __attribute__((ext_vector_type(8)));
typedef _Float16 h2   __attribute__((ext_vector_type(2)));
typedef float    v8f  __attribute__((ext_vector_type(8)));

#define M_TOT  8192
#define K_TOT  4096
#define N_TOT  11008
#define BM     128
#define BN     128
#define BK     64
#define NTILES (K_TOT / BK)   // 64
#define LDST   72             // LDS row stride in f16 (64 + 8 pad; 144B, 16B aligned)

union Frag16 { v16h v; v8h h[2]; };
union Pack8  { v8h  v; h2  p[4]; };
union HBits  { unsigned int u; h2 h; };

__global__ __launch_bounds__(256, 2)
void qlinear_w4a16_wmma(const _Float16* __restrict__ x,
                        const int*      __restrict__ qweight,
                        const unsigned int* __restrict__ qzeros,
                        const _Float16* __restrict__ scales,
                        const _Float16* __restrict__ bias,
                        _Float16*       __restrict__ out)
{
    __shared__ __align__(16) _Float16 As[2][BM * LDST];  // x tile, row-major (m, k)
    __shared__ __align__(16) _Float16 Bt[2][BN * LDST];  // dequantized W tile, transposed (n, k)

    const int tid  = threadIdx.x;
    const int lane = tid & 31;
    const int wid  = tid >> 5;       // 8 waves
    const int hl   = lane >> 4;      // lane half (0/1)
    const int l16  = lane & 15;

    const int n0 = blockIdx.x * BN;
    const int m0 = blockIdx.y * BM;

    // 2 (M) x 4 (N) wave grid; each wave owns 64x32 of the 128x128 tile.
    const int mW = (wid >> 2) * 64;
    const int nW = (wid & 3)  * 32;

    // --- dequant work mapping: thread -> (column nc, qweight row pair qr) ---
    const int nc     = tid & 127;          // column within tile
    const int qr     = tid >> 7;           // 0..1; rows handled: qr, qr+2, qr+4, qr+6
    const int nGlob  = n0 + nc;
    const int zShift = (nGlob & 7) * 4;
    const int zCol   = nGlob >> 3;

    // --- x load mapping: thread -> (row, 16-halfword chunk) ---
    const int xRow = tid >> 1;             // 0..127
    const int xOff = (tid & 1) * 16;       // 0 or 16 (f16 units)

    v8f acc[4][2];
#pragma unroll
    for (int i = 0; i < 4; ++i)
#pragma unroll
        for (int j = 0; j < 2; ++j)
            acc[i][j] = (v8f){};

    // Pipeline registers (globals for tile t+1 staged while computing tile t).
    v8h xr0, xr1;
    int qw[4];
    h2  zPk, sPk;

    auto loadG = [&](int kt) {
        const int k0   = kt * BK;
        const int g    = kt >> 1;          // 128-wide quant group
        const int qRow = k0 >> 3;

        const v8h* xg = (const v8h*)(x + (m0 + xRow) * K_TOT + k0 + xOff);
        xr0 = xg[0];
        xr1 = xg[1];

#pragma unroll
        for (int r = 0; r < 4; ++r)
            qw[r] = qweight[(qRow + qr + 2 * r) * N_TOT + nGlob];

        const unsigned int zn = (qzeros[g * (N_TOT / 8) + zCol] >> zShift) & 15u;
        HBits zb; zb.u = (0x6400u | zn) * 0x00010001u;   // packed fp16 (1024+z, 1024+z)
        zPk = zb.h;
        const _Float16 s = scales[g * N_TOT + nGlob];
        sPk = (h2){s, s};
    };

    auto stage = [&](int buf) {
        *(v8h*)&As[buf][xRow * LDST + xOff]     = xr0;
        *(v8h*)&As[buf][xRow * LDST + xOff + 8] = xr1;

#pragma unroll
        for (int r = 0; r < 4; ++r) {
            unsigned int q = (unsigned int)qw[r];
            Pack8 w;
#pragma unroll
            for (int p = 0; p < 4; ++p) {
                // halves = (1024 + nib_{2p}, 1024 + nib_{2p+1}) as fp16 bit patterns
                HBits t;
                t.u = ((q & 0x0000000Fu) | ((q << 12) & 0x000F0000u)) | 0x64006400u;
                w.p[p] = (t.h - zPk) * sPk;   // exact (nib - z), then fp16 * s
                q >>= 8;
            }
            *(v8h*)&Bt[buf][nc * LDST + (qr + 2 * r) * 8] = w.v;
        }
    };

    auto compute = [&](int buf) {
#pragma unroll
        for (int ks = 0; ks < 2; ++ks) {
            const int kk = ks * 32;

            Frag16 a[4];
#pragma unroll
            for (int i = 0; i < 4; ++i) {
                const _Float16* ap = &As[buf][(mW + i * 16 + l16) * LDST + kk];
                // ISA 16-bit A layout: VGPR0-3 = K[8*hl .. 8*hl+7],
                //                      VGPR4-7 = K[16+8*hl .. 16+8*hl+7]
                a[i].h[0] = *(const v8h*)(ap + 8 * hl);
                a[i].h[1] = *(const v8h*)(ap + 16 + 8 * hl);
            }

            Frag16 b[2];
#pragma unroll
            for (int j = 0; j < 2; ++j) {
                const _Float16* bp = &Bt[buf][(nW + j * 16 + l16) * LDST + kk + 8 * hl];
                b[j].h[0] = *(const v8h*)(bp);
                b[j].h[1] = *(const v8h*)(bp + 16);
            }

#pragma unroll
            for (int i = 0; i < 4; ++i)
#pragma unroll
                for (int j = 0; j < 2; ++j)
                    acc[i][j] = __builtin_amdgcn_wmma_f32_16x16x32_f16(
                        /*neg_a=*/false, a[i].v,
                        /*neg_b=*/false, b[j].v,
                        /*c_mod=*/(short)0, acc[i][j],
                        /*reuse_a=*/false, /*reuse_b=*/false);
        }
    };

    // ---- prologue: stage tile 0 ----
    loadG(0);
    stage(0);
    __syncthreads();

    // ---- steady state: load t+1 | compute t | stage t+1 | barrier ----
#pragma unroll 1
    for (int kt = 0; kt < NTILES - 1; ++kt) {
        loadG(kt + 1);
        compute(kt & 1);
        stage((kt + 1) & 1);
        __syncthreads();
    }
    compute((NTILES - 1) & 1);

    // ---- epilogue: C layout VGPR r -> row r + 8*hl, col = lane&15 ----
#pragma unroll
    for (int j = 0; j < 2; ++j) {
        const int      ng = n0 + nW + j * 16 + l16;
        const _Float16 bv = bias[ng];
#pragma unroll
        for (int i = 0; i < 4; ++i) {
            const int mrow = m0 + mW + i * 16 + hl * 8;
#pragma unroll
            for (int r = 0; r < 8; ++r)
                out[(mrow + r) * N_TOT + ng] = (_Float16)acc[i][j][r] + bv;
        }
    }
}

extern "C" void kernel_launch(void* const* d_in, const int* in_sizes, int n_in,
                              void* d_out, int out_size, void* d_ws, size_t ws_size,
                              hipStream_t stream)
{
    const _Float16*     x       = (const _Float16*)d_in[0];
    const int*          qweight = (const int*)d_in[1];
    const unsigned int* qzeros  = (const unsigned int*)d_in[2];
    const _Float16*     scales  = (const _Float16*)d_in[3];
    // d_in[4] = g_idx: groups are contiguous (k / 128); recomputed inline.
    const _Float16*     bias    = (const _Float16*)d_in[5];
    _Float16*           out     = (_Float16*)d_out;

    dim3 grid(N_TOT / BN, M_TOT / BM);   // 86 x 64 workgroups
    dim3 block(256);                     // 8 wave32s
    qlinear_w4a16_wmma<<<grid, block, 0, stream>>>(x, qweight, qzeros, scales, bias, out);
}
